// QKVPackedGQA_14834817040863
// MI455X (gfx1250) — matrix operand order repaired and checked by reading, not compiled, not run on hardware
//
#include <hip/hip_runtime.h>

// ---------------------------------------------------------------------------
// Fused GQA attention block for MI455X (gfx1250, wave32, WMMA).
//   qkv = x @ wqkv^T ; rope(q,k) ; flash-attn (causal, GQA 4:1) ; out = ctx @ wo^T
// GEMM-shaped math on v_wmma_f32_16x16x32_bf16; both GEMMs and the attention
// K/V stream are staged through LDS with double-buffered
// global_load_async_to_lds_b128 (ASYNCcnt) copies.
// ---------------------------------------------------------------------------

typedef __attribute__((ext_vector_type(16))) __bf16 v16bf;
typedef __attribute__((ext_vector_type(8)))  float  v8f;

union BFrag {
    v16bf v;
    uint4 q[2];
    unsigned short us[16];
};

__device__ __forceinline__ unsigned short f2bf(float f) {
    unsigned u = __float_as_uint(f);
    u += 0x7FFFu + ((u >> 16) & 1u);          // round-to-nearest-even
    return (unsigned short)(u >> 16);
}
__device__ __forceinline__ float bf2f(unsigned short h) {
    return __uint_as_float(((unsigned)h) << 16);
}

// Async global->LDS copy, 16B per lane (VGLOBAL opcode 98, tracked by ASYNCcnt).
__device__ __forceinline__ void async_b128(const void* smem_dst, const void* gsrc) {
    unsigned lds_off = (unsigned)(unsigned long long)smem_dst;  // addr[31:0] = LDS offset
    asm volatile("global_load_async_to_lds_b128 %0, %1, off"
                 :: "v"(lds_off), "v"((unsigned long long)gsrc)
                 : "memory");
}
#define WAIT_ASYNCCNT(n) asm volatile("s_wait_asynccnt " #n ::: "memory")

// ------------------------------- constants ---------------------------------
#define BATCH   2
#define SEQ     2048
#define EMBED   4096
#define NHEADS  32
#define NKVH    8
#define HDIM    128
#define FDIM    6144                    // EMBED + 2*KV_DIM
#define SCALE   0.08838834764831845f    // 1/sqrt(128)
#define LDK     136                     // padded LDS row stride (elems), attn K
#define LDV     40                      // padded LDS row stride (elems), attn V^T
#define LDT     40                      // padded LDS row stride (elems), GEMM tiles

// --------------------------- fp32 -> bf16 convert --------------------------
__global__ void convert_f32_bf16(const float* __restrict__ in,
                                 unsigned short* __restrict__ out, size_t n) {
    size_t i = (size_t)blockIdx.x * blockDim.x + threadIdx.x;
    size_t stride = (size_t)gridDim.x * blockDim.x;
    for (; i < n; i += stride) out[i] = f2bf(in[i]);
}

// ----------------------------- WMMA GEMM -----------------------------------
// C[M][N] = A[M][K] (bf16 row-major) * W[N][K]^T (bf16 rows = N).
// Block = 256 threads (8 waves).  Block tile 64(M) x 256(N), k-step 32.
// A (64x32) and B (256x32) tiles staged in LDS by double-buffered async
// copies; wave tile 32x64 = 2 A-frags x 4 B-frags -> 8 WMMAs per k-step.
template <bool OUT_BF16>
__global__ __launch_bounds__(256)
void gemm_wmma_bf16(const unsigned short* __restrict__ A,
                    const unsigned short* __restrict__ W,
                    void* __restrict__ Cout, int M, int N, int K) {
    __shared__ __align__(16) unsigned short sA[2][64 * LDT];   // 2 x 5 KB
    __shared__ __align__(16) unsigned short sB[2][256 * LDT];  // 2 x 20 KB

    const int tid  = threadIdx.x;
    const int wave = tid >> 5;
    const int lane = tid & 31;
    const int half = lane >> 4;
    const int l    = lane & 15;
    const int mw = (wave >> 2) * 32;           // wave tile inside block tile
    const int nw = (wave & 3) * 64;
    const int m0 = blockIdx.x * 64;
    const int n0 = blockIdx.y * 256;

    const unsigned short* abase = A + (size_t)m0 * K;
    const unsigned short* wbase = W + (size_t)n0 * K;

    v8f acc[2][4];
#pragma unroll
    for (int i = 0; i < 2; ++i)
#pragma unroll
        for (int j = 0; j < 4; ++j)
#pragma unroll
            for (int r = 0; r < 8; ++r) acc[i][j][r] = 0.0f;

    // ---- cooperative async copy of one k-chunk into buffer bi ----
    auto issue_chunk = [&](int k0, int bi) {
#pragma unroll
        for (int j = 0; j < 2; ++j) {          // B: 256 rows x 32 elems (512 groups)
            const int g = j * 256 + tid;
            const int row = g >> 1, c = g & 1;
            async_b128(&sB[bi][row * LDT + c * 16],
                       wbase + (size_t)row * K + k0 + c * 16);
        }
        if (tid < 128) {                       // A: 64 rows x 32 elems (128 groups)
            const int row = tid >> 1, c = tid & 1;
            async_b128(&sA[bi][row * LDT + c * 16],
                       abase + (size_t)row * K + k0 + c * 16);
        }
    };

    issue_chunk(0, 0);
    const int nsteps = K / 32;

    for (int ks = 0; ks < nsteps; ++ks) {
        const int buf = ks & 1;
        if (ks + 1 < nsteps) {
            issue_chunk((ks + 1) * 32, buf ^ 1);
            if (wave < 4) { WAIT_ASYNCCNT(3); } else { WAIT_ASYNCCNT(2); }
        } else {
            WAIT_ASYNCCNT(0);
        }
        __syncthreads();                       // all waves' copies visible

        BFrag a0, a1, b[4];
        {   // A fragments: lane<16 holds K 0..15 of its row.
            const uint4* p = (const uint4*)&sA[buf][(mw + l) * LDT + 16 * half];
            a0.q[0] = p[0]; a0.q[1] = p[1];
            p = (const uint4*)&sA[buf][(mw + 16 + l) * LDT + 16 * half];
            a1.q[0] = p[0]; a1.q[1] = p[1];
        }
#pragma unroll
        for (int j = 0; j < 4; ++j) {          // B fragments: 32 contiguous elems
            const uint4* p =
                (const uint4*)&sB[buf][(nw + j * 16 + l) * LDT + 16 * half];
            b[j].q[0] = p[0]; b[j].q[1] = p[1];
        }
#pragma unroll
        for (int j = 0; j < 4; ++j)
            acc[0][j] = __builtin_amdgcn_wmma_f32_16x16x32_bf16(
                false, a0.v, false, b[j].v, (short)0, acc[0][j], false, false);
#pragma unroll
        for (int j = 0; j < 4; ++j)
            acc[1][j] = __builtin_amdgcn_wmma_f32_16x16x32_bf16(
                false, a1.v, false, b[j].v, (short)0, acc[1][j], false, false);

        __syncthreads();                       // before next chunk overwrites buf
    }

#pragma unroll
    for (int i = 0; i < 2; ++i)
#pragma unroll
        for (int j = 0; j < 4; ++j)
#pragma unroll
            for (int r = 0; r < 8; ++r) {
                size_t idx = (size_t)(m0 + mw + 16 * i + r + 8 * half) * N +
                             n0 + nw + j * 16 + l;
                if (OUT_BF16) ((unsigned short*)Cout)[idx] = f2bf(acc[i][j][r]);
                else          ((float*)Cout)[idx]          = acc[i][j][r];
            }
}

// ------------------------ RoPE + GQA scatter -------------------------------
// qkv_bf: [B*S][6144] where f = kvh*768 + slot*128 + d  (slots 0..3=q, 4=k, 5=v)
__global__ void rope_scatter(const unsigned short* __restrict__ qkv,
                             unsigned short* __restrict__ Qo,
                             unsigned short* __restrict__ Ko,
                             unsigned short* __restrict__ Vt) {
    const int rid = blockIdx.x;
    const int d   = threadIdx.x;                // 0..127
    const int r     = rid % (NHEADS + 2 * NKVH);
    const int token = rid / (NHEADS + 2 * NKVH);
    const int b = token / SEQ;
    const int s = token % SEQ;
    const unsigned short* row = qkv + (size_t)token * FDIM;

    if (r < NHEADS) {                           // query head, RoPE
        const int h = r, kvh = h >> 2, slot = h & 3;
        const int fb = kvh * 768 + slot * 128;
        const int i  = d & 63;
        const float inv = __expf(-(float)(2 * i) * (9.210340371976184f / 128.0f));
        const float ang = (float)s * inv;
        const float c = __cosf(ang), sn = __sinf(ang);
        const float xv = bf2f(row[fb + d]);
        const float xr = (d < 64) ? -bf2f(row[fb + d + 64]) : bf2f(row[fb + d - 64]);
        Qo[((size_t)(b * NHEADS + h) * SEQ + s) * HDIM + d] = f2bf(xv * c + xr * sn);
    } else if (r < NHEADS + NKVH) {             // key head, RoPE
        const int kvh = r - NHEADS;
        const int fb = kvh * 768 + 4 * 128;
        const int i  = d & 63;
        const float inv = __expf(-(float)(2 * i) * (9.210340371976184f / 128.0f));
        const float ang = (float)s * inv;
        const float c = __cosf(ang), sn = __sinf(ang);
        const float xv = bf2f(row[fb + d]);
        const float xr = (d < 64) ? -bf2f(row[fb + d + 64]) : bf2f(row[fb + d - 64]);
        Ko[((size_t)(b * NKVH + kvh) * SEQ + s) * HDIM + d] = f2bf(xv * c + xr * sn);
    } else {                                    // value head -> V^T [D][S]
        const int kvh = r - NHEADS - NKVH;
        const int fb = kvh * 768 + 5 * 128;
        Vt[((size_t)(b * NKVH + kvh) * HDIM + d) * SEQ + s] = row[fb + d];
    }
}

// ------------------------ flash attention (transposed) ---------------------
// Block = 128 threads (4 waves) = one head x 64 queries.  Sᵀ = K·Qᵀ keeps the
// query dim on lanes (softmax = 8 regs + one xor-16 shuffle; Pᵀ -> B-frag via
// one xor-16 shuffle, zero LDS transposes).  K/V chunks (32 kpos) are staged
// in LDS once per block by double-buffered async global->LDS copies.
__global__ __launch_bounds__(128)
void flash_attn_wmma(const unsigned short* __restrict__ Q,
                     const unsigned short* __restrict__ Kc,
                     const unsigned short* __restrict__ Vt,
                     unsigned short* __restrict__ Ctx) {
    __shared__ __align__(16) unsigned short sK[2][32 * LDK];   // 17.4 KB
    __shared__ __align__(16) unsigned short sV[2][HDIM * LDV]; // 20.5 KB

    const int tid  = threadIdx.x;
    const int wave = tid >> 5;
    const int lane = tid & 31;
    const int half = lane >> 4;
    const int l    = lane & 15;
    const int QT = SEQ / 64;
    const int qt = blockIdx.x % QT;
    const int h  = (blockIdx.x / QT) % NHEADS;
    const int b  = blockIdx.x / (QT * NHEADS);
    const int kvh = h >> 2;
    const int q0  = qt * 64 + wave * 16;

    const unsigned short* qb = Q  + (size_t)(b * NHEADS + h) * SEQ * HDIM;
    const unsigned short* kb = Kc + (size_t)(b * NKVH + kvh) * SEQ * HDIM;
    const unsigned short* vb = Vt + (size_t)(b * NKVH + kvh) * HDIM * SEQ;

    // Qᵀ as 4 B-fragments (d-chunks of 32), kept in registers.
    BFrag qf[4];
#pragma unroll
    for (int t = 0; t < 4; ++t) {
        const uint4* p =
            (const uint4*)(qb + (size_t)(q0 + l) * HDIM + 32 * t + 8 * half);
        qf[t].q[0] = p[0];
        qf[t].q[1] = p[2];
    }

    v8f o[8];
#pragma unroll
    for (int t = 0; t < 8; ++t)
#pragma unroll
        for (int r = 0; r < 8; ++r) o[t][r] = 0.0f;

    float mrow = -3.0e38f, lrow = 0.0f;
    const int qpos = q0 + l;
    const int nmax = qt * 2 + 2;    // block-uniform chunk count (extra chunks
                                    // for low waves are fully masked: p==0)

    // ---- issue async copies of chunk kc into buffer bi (6 instrs / wave) ----
    auto issue_chunk = [&](int kc, int bi) {
        const int k0 = kc * 32;
#pragma unroll
        for (int j = 0; j < 4; ++j) {          // K: 32 rows x 128 elems
            const int g = j * 128 + tid;       // 512 groups of 16 elems
            const int row = g >> 3, c = g & 7;
            async_b128(&sK[bi][row * LDK + c * 16],
                       kb + (size_t)(k0 + row) * HDIM + c * 16);
        }
#pragma unroll
        for (int j = 0; j < 2; ++j) {          // Vᵀ: 128 rows x 32 elems
            const int g = j * 128 + tid;       // 256 groups of 16 elems
            const int d = g >> 1, c = g & 1;
            async_b128(&sV[bi][d * LDV + c * 16],
                       vb + (size_t)d * SEQ + k0 + c * 16);
        }
    };

    issue_chunk(0, 0);

    for (int kc = 0; kc < nmax; ++kc) {
        const int k0  = kc * 32;
        const int buf = kc & 1;
        if (kc + 1 < nmax) {
            issue_chunk(kc + 1, buf ^ 1);
            WAIT_ASYNCCNT(6);                  // chunk kc's copies are done
        } else {
            WAIT_ASYNCCNT(0);
        }
        __syncthreads();                       // all waves' copies visible

        v8f s0, s1;
#pragma unroll
        for (int r = 0; r < 8; ++r) { s0[r] = 0.0f; s1[r] = 0.0f; }

        BFrag kf[4];
#pragma unroll
        for (int t = 0; t < 4; ++t) {          // Sᵀ[k0..k0+15][q]
            const uint4* p = (const uint4*)&sK[buf][l * LDK + 32 * t + 8 * half];
            kf[t].q[0] = p[0]; kf[t].q[1] = p[2];
        }
#pragma unroll
        for (int t = 0; t < 4; ++t)
            s0 = __builtin_amdgcn_wmma_f32_16x16x32_bf16(
                false, kf[t].v, false, qf[t].v, (short)0, s0, false, false);
#pragma unroll
        for (int t = 0; t < 4; ++t) {          // Sᵀ[k0+16..k0+31][q]
            const uint4* p =
                (const uint4*)&sK[buf][(16 + l) * LDK + 32 * t + 8 * half];
            kf[t].q[0] = p[0]; kf[t].q[1] = p[2];
        }
#pragma unroll
        for (int t = 0; t < 4; ++t)
            s1 = __builtin_amdgcn_wmma_f32_16x16x32_bf16(
                false, kf[t].v, false, qf[t].v, (short)0, s1, false, false);

        // ------- online softmax over this 32-wide k chunk -------
        float p0[8], p1[8];
        float mloc = -3.0e38f;
#pragma unroll
        for (int r = 0; r < 8; ++r) {
            const int kp0 = k0 + r + 8 * half;
            const int kp1 = kp0 + 16;
            float a0 = (kp0 <= qpos) ? s0[r] * SCALE : -3.0e38f;
            float a1 = (kp1 <= qpos) ? s1[r] * SCALE : -3.0e38f;
            p0[r] = a0; p1[r] = a1;
            mloc = fmaxf(mloc, fmaxf(a0, a1));
        }
        mloc = fmaxf(mloc, __shfl_xor(mloc, 16, 32));
        const float mnew  = fmaxf(mrow, mloc);
        const float alpha = __expf(mrow - mnew);
        float ps = 0.0f;
#pragma unroll
        for (int r = 0; r < 8; ++r) {
            p0[r] = __expf(p0[r] - mnew);
            p1[r] = __expf(p1[r] - mnew);
            ps += p0[r] + p1[r];
        }
        ps += __shfl_xor(ps, 16, 32);
        lrow = lrow * alpha + ps;
        mrow = mnew;
#pragma unroll
        for (int t = 0; t < 8; ++t)
#pragma unroll
            for (int r = 0; r < 8; ++r) o[t][r] *= alpha;

        // ------- Pᵀ (32x16) as B-fragment via one xor-16 shuffle -------
        BFrag pb;
#pragma unroll
        for (int r = 0; r < 8; ++r) {
            const float o0 = __shfl_xor(p0[r], 16, 32);
            const float o1 = __shfl_xor(p1[r], 16, 32);
            if (half == 0) { pb.us[r] = f2bf(p0[r]); pb.us[8 + r] = f2bf(o0); }
            else           { pb.us[r] = f2bf(o1);    pb.us[8 + r] = f2bf(p1[r]); }
        }

        // ------- ctxᵀ += Vᵀ · Pᵀ  (8 d-tiles, V frags from LDS) -------
        BFrag vf[4];
#pragma unroll
        for (int t = 0; t < 4; ++t) {
            const uint4* p =
                (const uint4*)&sV[buf][(16 * t + l) * LDV + 16 * half];
            vf[t].q[0] = p[0]; vf[t].q[1] = p[1];
        }
#pragma unroll
        for (int t = 0; t < 4; ++t)
            o[t] = __builtin_amdgcn_wmma_f32_16x16x32_bf16(
                false, vf[t].v, false, pb.v, (short)0, o[t], false, false);
#pragma unroll
        for (int t = 0; t < 4; ++t) {
            const uint4* p =
                (const uint4*)&sV[buf][(64 + 16 * t + l) * LDV + 16 * half];
            vf[t].q[0] = p[0]; vf[t].q[1] = p[1];
        }
#pragma unroll
        for (int t = 0; t < 4; ++t)
            o[4 + t] = __builtin_amdgcn_wmma_f32_16x16x32_bf16(
                false, vf[t].v, false, pb.v, (short)0, o[4 + t], false, false);

        __syncthreads();                       // before next chunk overwrites buf
    }

    const float inv = 1.0f / lrow;
#pragma unroll
    for (int t = 0; t < 8; ++t) {
        uint4 st;
        unsigned short* sp = (unsigned short*)&st;
#pragma unroll
        for (int r = 0; r < 8; ++r) sp[r] = f2bf(o[t][r] * inv);
        size_t idx = (size_t)(b * SEQ + q0 + l) * (NHEADS * HDIM) +
                     h * HDIM + 16 * t + 8 * half;
        *(uint4*)(Ctx + idx) = st;
    }
}

// ------------------------------- launcher ----------------------------------
extern "C" void kernel_launch(void* const* d_in, const int* in_sizes, int n_in,
                              void* d_out, int out_size, void* d_ws, size_t ws_size,
                              hipStream_t stream) {
    const float* x    = (const float*)d_in[0];   // [2,2048,4096]
    const float* wqkv = (const float*)d_in[1];   // [6144,4096]
    const float* wo   = (const float*)d_in[2];   // [4096,4096]
    float* out = (float*)d_out;                  // [2,2048,4096]

    const size_t NX    = (size_t)BATCH * SEQ * EMBED;
    const size_t NWQKV = (size_t)FDIM * EMBED;
    const size_t NWO   = (size_t)EMBED * EMBED;

    char* ws = (char*)d_ws;
    size_t off = 0;
    unsigned short* x_bf    = (unsigned short*)(ws + off); off += NX * 2;
    unsigned short* wqkv_bf = (unsigned short*)(ws + off); off += NWQKV * 2;
    unsigned short* wo_bf   = (unsigned short*)(ws + off); off += NWO * 2;
    unsigned short* qkv_bf  = (unsigned short*)(ws + off); off += (size_t)BATCH * SEQ * FDIM * 2;
    unsigned short* q_bf    = (unsigned short*)(ws + off); off += (size_t)BATCH * NHEADS * SEQ * HDIM * 2;
    unsigned short* k_bf    = (unsigned short*)(ws + off); off += (size_t)BATCH * NKVH * SEQ * HDIM * 2;
    unsigned short* v_bf    = (unsigned short*)(ws + off); off += (size_t)BATCH * NKVH * HDIM * SEQ * 2;
    unsigned short* ctx_bf  = (unsigned short*)(ws + off); off += NX * 2;
    // total workspace used: ~252 MB

    convert_f32_bf16<<<4096, 256, 0, stream>>>(x,    x_bf,    NX);
    convert_f32_bf16<<<4096, 256, 0, stream>>>(wqkv, wqkv_bf, NWQKV);
    convert_f32_bf16<<<4096, 256, 0, stream>>>(wo,   wo_bf,   NWO);

    {   // QKV projection: [4096 x 6144]
        dim3 grid((BATCH * SEQ) / 64, FDIM / 256);
        gemm_wmma_bf16<true><<<grid, 256, 0, stream>>>(
            x_bf, wqkv_bf, qkv_bf, BATCH * SEQ, FDIM, EMBED);
    }

    rope_scatter<<<BATCH * SEQ * (NHEADS + 2 * NKVH), HDIM, 0, stream>>>(
        qkv_bf, q_bf, k_bf, v_bf);

    flash_attn_wmma<<<BATCH * NHEADS * (SEQ / 64), 128, 0, stream>>>(
        q_bf, k_bf, v_bf, ctx_bf);

    {   // output projection: [4096 x 4096], fp32 out
        dim3 grid((BATCH * SEQ) / 64, EMBED / 256);
        gemm_wmma_bf16<false><<<grid, 256, 0, stream>>>(
            ctx_bf, wo_bf, (void*)out, BATCH * SEQ, EMBED, EMBED);
    }
}
